// ParallelExperts_22651657519433
// MI455X (gfx1250) — matrix operand order repaired and checked by reading, not compiled, not run on hardware
//
#include <hip/hip_runtime.h>
#include <hip/hip_bf16.h>
#include <math.h>

// ---------------------------------------------------------------------------
// MoE expert-choice layer for MI455X (gfx1250), wave32 + WMMA bf16 + TDM.
//   N=8192 tokens, C=1024, E=32 experts, D=2048, k=1024 tokens/expert.
//   275 GFLOP of GEMM -> compute-bound -> v_wmma_f32_16x16x32_bf16,
//   double-buffered LDS, TDM async tensor load for the bf16 A operand.
// ---------------------------------------------------------------------------

typedef __bf16 bf16_t;
typedef __attribute__((ext_vector_type(16))) __bf16 v16bf;
typedef __attribute__((ext_vector_type(8)))  __bf16 v8bf;
typedef __attribute__((ext_vector_type(4)))  __bf16 v4bf;
typedef __attribute__((ext_vector_type(8)))  float  v8f;
typedef __attribute__((ext_vector_type(4)))  float  v4f;
typedef __attribute__((ext_vector_type(4)))  unsigned int u32x4;
typedef __attribute__((ext_vector_type(8)))  int    i32x8;
typedef __attribute__((ext_vector_type(4)))  int    i32x4;

#define N_TOK 8192
#define C_DIM 1024
#define E_EXP 32
#define D_DIM 2048
#define K_SEL 1024   // tokens chosen per expert (N / EXPANSION)

#if defined(__has_builtin)
#  if __has_builtin(__builtin_amdgcn_tensor_load_to_lds) && __has_builtin(__builtin_amdgcn_s_wait_tensorcnt)
#    define USE_TDM 1
#  else
#    define USE_TDM 0
#  endif
#else
#  define USE_TDM 0
#endif

// ---------------------------------------------------------------------------
// Kernel 1: router logits + softmax weights; also zero fanout (d_out poisoned)
// ---------------------------------------------------------------------------
__global__ __launch_bounds__(256) void router_kernel(
    const float* __restrict__ x, const float* __restrict__ Wr,
    float* __restrict__ logits, float* __restrict__ weights,
    float* __restrict__ fanout)
{
  __shared__ __align__(16) float xs[C_DIM];
  __shared__ float part[8][E_EXP];
  const int n = blockIdx.x;
  const int t = threadIdx.x;

  ((v4f*)xs)[t] = ((const v4f*)(x + (size_t)n * C_DIM))[t];
  if (t == 0) fanout[n] = 0.0f;
  __syncthreads();

  const int e   = t & (E_EXP - 1);
  const int seg = t >> 5;
  const float* wrow = Wr + (size_t)e * C_DIM + seg * 128;
  const float* xseg = xs + seg * 128;
  float acc = 0.f;
  #pragma unroll 8
  for (int c = 0; c < 128; ++c) acc = fmaf(xseg[c], wrow[c], acc);
  part[seg][e] = acc;
  __syncthreads();

  if (t < E_EXP) {
    float l = 0.f;
    #pragma unroll
    for (int s = 0; s < 8; ++s) l += part[s][t];
    logits[(size_t)n * E_EXP + t] = l;
    float m = l;
    #pragma unroll
    for (int off = 16; off > 0; off >>= 1) m = fmaxf(m, __shfl_xor(m, off, 32));
    float ex = __expf(l - m);
    float s = ex;
    #pragma unroll
    for (int off = 16; off > 0; off >>= 1) s += __shfl_xor(s, off, 32);
    weights[(size_t)n * E_EXP + t] = ex / s;
  }
}

// ---------------------------------------------------------------------------
// Kernel 2: per-expert top-k over tokens via full bitonic sort in LDS.
// Key: value desc, tie -> index asc (matches jax.lax.top_k ordering).
// ---------------------------------------------------------------------------
__global__ __launch_bounds__(1024) void topk_kernel(
    const float* __restrict__ logits, const float* __restrict__ weights,
    int* __restrict__ topk_idx, float* __restrict__ out_li,
    float* __restrict__ out_w, float* __restrict__ fanout)
{
  __shared__ float vv[N_TOK];
  __shared__ int   id[N_TOK];
  const int e = blockIdx.x;
  const int t = threadIdx.x;

  #pragma unroll
  for (int q = 0; q < N_TOK / 1024; ++q) {
    int i = t + q * 1024;
    vv[i] = logits[(size_t)i * E_EXP + e];
    id[i] = i;
  }
  __syncthreads();

  for (int k = 2; k <= N_TOK; k <<= 1) {
    for (int j = k >> 1; j > 0; j >>= 1) {
      #pragma unroll
      for (int q = 0; q < N_TOK / 1024; ++q) {
        int i = t + q * 1024;
        int l = i ^ j;
        if (l > i) {
          float vi = vv[i], vl = vv[l];
          int   ii = id[i], il = id[l];
          bool l_before_i = (vl > vi) || (vl == vi && il < ii);
          bool doswap = ((i & k) == 0) ? l_before_i : !l_before_i;
          if (doswap) { vv[i] = vl; vv[l] = vi; id[i] = il; id[l] = ii; }
        }
      }
      __syncthreads();
    }
  }

  if (t < K_SEL) {
    int idx = id[t];
    int g = e * K_SEL + t;
    topk_idx[g] = idx;
    out_li[g]   = (float)idx;
    out_w[g]    = weights[(size_t)idx * E_EXP + e];
    atomicAdd(&fanout[idx], 1.0f);
  }
}

// ---------------------------------------------------------------------------
// WMMA fragment loaders (ISA 7.12.2 16-bit layouts, wave32)
// ---------------------------------------------------------------------------
__device__ __attribute__((always_inline)) inline v16bf load_a_frag(const bf16_t* As, int mbase) {
  const int lane = threadIdx.x & 31;
  const bf16_t* p = As + (mbase + (lane & 15)) * 32 + (lane >> 4) * 8;
  v8bf lo = *(const v8bf*)(p);
  v8bf hi = *(const v8bf*)(p + 16);
  return __builtin_shufflevector(lo, hi, 0,1,2,3,4,5,6,7,8,9,10,11,12,13,14,15);
}
__device__ __attribute__((always_inline)) inline v16bf load_b_frag(const bf16_t* Bs, int nbase) {
  const int lane = threadIdx.x & 31;
  const bf16_t* p = Bs + (nbase + (lane & 15)) * 32 + (lane >> 4) * 8;
  v8bf lo = *(const v8bf*)(p);
  v8bf hi = *(const v8bf*)(p + 16);
  return __builtin_shufflevector(lo, hi, 0,1,2,3,4,5,6,7,8,9,10,11,12,13,14,15);
}

__device__ __attribute__((always_inline)) inline
void mma_step(const bf16_t* As, const bf16_t* Bs, int wm, int wn, v8f (&acc)[4][2]) {
  v16bf af[4], bfr[2];
  #pragma unroll
  for (int i = 0; i < 4; ++i) af[i] = load_a_frag(As, wm + i * 16);
  #pragma unroll
  for (int j = 0; j < 2; ++j) bfr[j] = load_b_frag(Bs, wn + j * 16);
  #pragma unroll
  for (int i = 0; i < 4; ++i)
    #pragma unroll
    for (int j = 0; j < 2; ++j)
      acc[i][j] = __builtin_amdgcn_wmma_f32_16x16x32_bf16(
          false, af[i], false, bfr[j], (short)0, acc[i][j], false, false);
}

// Branch-free tanh-GELU: 0.5v(1+tanh(u)) == v * sigmoid(2u)
__device__ __attribute__((always_inline)) inline float gelu_fast(float v) {
  float t2 = 1.5957691216057308f * v * fmaf(0.044715f, v * v, 1.0f);
  return v * __builtin_amdgcn_rcpf(1.0f + __expf(-t2));
}

// f32x4 -> bf16x4 LDS stage
__device__ __attribute__((always_inline)) inline void cvt_store4(bf16_t* dst, v4f a) {
  v4bf p = { (bf16_t)a.x, (bf16_t)a.y, (bf16_t)a.z, (bf16_t)a.w };
  *(v4bf*)dst = p;
}

#if USE_TDM
// ---------------------------------------------------------------------------
// TDM: async 2D tensor load (bf16 128x32 tile, row stride D_DIM) into LDS.
// D# per ISA 8.3/8.4: group0 {count=1, lds_addr, global_addr, type=2},
// group1 {data_size=2B, tensor 2048x1024, tile 32x128, stride 2048}.
// ---------------------------------------------------------------------------
__device__ __attribute__((always_inline)) inline
void tdm_load_tile_a(const bf16_t* gsrc, bf16_t* ldst) {
  u32x4 g0;
  g0.x = 1u;                                        // count=1, user mode
  g0.y = (unsigned)(uintptr_t)ldst;                 // LDS byte addr = addr[31:0]
  unsigned long long ga = (unsigned long long)(uintptr_t)gsrc;
  g0.z = (unsigned)ga;                              // global_addr[31:0]
  g0.w = (unsigned)((ga >> 32) & 0x1ffffffull) | (2u << 30);  // addr[56:32] | type=2
  i32x8 g1;
  g1[0] = 0x00010000;                 // workgroup_mask=0, data_size=1 (2 bytes)
  g1[1] = (int)(2048u << 16);         // tensor_dim0 = 2048 (bits 63:48)
  g1[2] = (int)(1024u << 16);         // tensor_dim1 = 1024 (bits 95:80)
  g1[3] = (int)(32u   << 16);         // tile_dim0 = 32     (bits 127:112)
  g1[4] = 128;                        // tile_dim1 = 128, tile_dim2 = 0
  g1[5] = 2048;                       // tensor_dim0_stride = D_DIM
  g1[6] = 0;
  g1[7] = 0;
  i32x4 z4 = {0, 0, 0, 0};
#if __clang_major__ >= 23
  i32x8 z8 = {0, 0, 0, 0, 0, 0, 0, 0};
  __builtin_amdgcn_tensor_load_to_lds(g0, g1, z4, z4, z8, 0);
#else
  __builtin_amdgcn_tensor_load_to_lds(g0, g1, z4, z4, 0);
#endif
}
#endif

// ---------------------------------------------------------------------------
// Kernel 3: hidden = gelu(gather(x, idx) @ W1[e]^T), bf16 out.
// 128x128x32 tiles, double-buffered LDS, 8 waves -> 4x2 wmma accs each.
// ---------------------------------------------------------------------------
__global__ __launch_bounds__(256) void expert_gemm1(
    const float* __restrict__ x, const float* __restrict__ W1,
    const int* __restrict__ topk_idx, bf16_t* __restrict__ hidden)
{
  __shared__ __align__(64) bf16_t As[2][128 * 32];
  __shared__ __align__(64) bf16_t Bs[2][128 * 32];
  __shared__ int rowidx[128];

  const int e  = blockIdx.z;
  const int m0 = blockIdx.y * 128;
  const int n0 = blockIdx.x * 128;
  const int t  = threadIdx.x;
  const int wave = t >> 5, lane = t & 31;
  const int wm = (wave & 1) * 64;
  const int wn = (wave >> 1) * 32;

  if (t < 128) rowidx[t] = topk_idx[e * K_SEL + m0 + t];
  __syncthreads();

  // per-thread staging slots (fixed rows across the K loop)
  const float* asrc[4];
  const float* bsrc[4];
  int scol[4], srow[4];
  {
    const float* Wb = W1 + (size_t)e * D_DIM * C_DIM;
    #pragma unroll
    for (int q = 0; q < 4; ++q) {
      int chunk = t + q * 256;
      srow[q] = chunk >> 3;
      scol[q] = (chunk & 7) * 4;
      asrc[q] = x  + (size_t)rowidx[srow[q]] * C_DIM + scol[q];
      bsrc[q] = Wb + (size_t)(n0 + srow[q]) * C_DIM + scol[q];
    }
  }

  v8f acc[4][2];
  {
    v8f z = {0.f,0.f,0.f,0.f,0.f,0.f,0.f,0.f};
    #pragma unroll
    for (int i = 0; i < 4; ++i)
      #pragma unroll
      for (int j = 0; j < 2; ++j) acc[i][j] = z;
  }

  v4f ar[4], br[4];
  #pragma unroll
  for (int q = 0; q < 4; ++q) { ar[q] = *(const v4f*)(asrc[q]); br[q] = *(const v4f*)(bsrc[q]); }
  #pragma unroll
  for (int q = 0; q < 4; ++q) {
    cvt_store4(As[0] + srow[q] * 32 + scol[q], ar[q]);
    cvt_store4(Bs[0] + srow[q] * 32 + scol[q], br[q]);
  }

  const int KT = C_DIM / 32;
  int cur = 0;
  for (int kt = 0; kt < KT; ++kt) {
    __syncthreads();
    const bool more = (kt + 1 < KT);
    if (more) {
      int k0n = (kt + 1) * 32;
      #pragma unroll
      for (int q = 0; q < 4; ++q) { ar[q] = *(const v4f*)(asrc[q] + k0n); br[q] = *(const v4f*)(bsrc[q] + k0n); }
    }
    mma_step(As[cur], Bs[cur], wm, wn, acc);
    if (more) {
      #pragma unroll
      for (int q = 0; q < 4; ++q) {
        cvt_store4(As[cur ^ 1] + srow[q] * 32 + scol[q], ar[q]);
        cvt_store4(Bs[cur ^ 1] + srow[q] * 32 + scol[q], br[q]);
      }
    }
    cur ^= 1;
  }

  const int rhalf = (lane >> 4) * 8;
  #pragma unroll
  for (int i = 0; i < 4; ++i) {
    #pragma unroll
    for (int j = 0; j < 2; ++j) {
      int col = n0 + wn + j * 16 + (lane & 15);
      #pragma unroll
      for (int p = 0; p < 8; ++p) {
        int row = m0 + wm + i * 16 + rhalf + p;
        hidden[((size_t)e * K_SEL + row) * D_DIM + col] = (bf16_t)gelu_fast(acc[i][j][p]);
      }
    }
  }
}

// ---------------------------------------------------------------------------
// Kernel 4: h = hidden @ W2[e]^T, f32 out. A (bf16) tiles fetched by the
// Tensor Data Mover into LDS (wave 0 issues, s_wait_tensorcnt + barrier
// publishes); B converted f32->bf16 by the block. Double buffered.
// ---------------------------------------------------------------------------
__global__ __launch_bounds__(256) void expert_gemm2(
    const bf16_t* __restrict__ hidden, const float* __restrict__ W2,
    float* __restrict__ h_flat)
{
  __shared__ __align__(64) bf16_t As[2][128 * 32];
  __shared__ __align__(64) bf16_t Bs[2][128 * 32];

  const int e  = blockIdx.z;
  const int m0 = blockIdx.y * 128;
  const int n0 = blockIdx.x * 128;
  const int t  = threadIdx.x;
  const int wave = t >> 5, lane = t & 31;
  const int wm = (wave & 1) * 64;
  const int wn = (wave >> 1) * 32;

  const bf16_t* Ab = hidden + ((size_t)e * K_SEL + m0) * D_DIM;
  const float*  Wb = W2 + (size_t)e * C_DIM * D_DIM;

  const float* bsrc[4];
#if !USE_TDM
  const bf16_t* a2src[4];
#endif
  int scol[4], srow[4];
  #pragma unroll
  for (int q = 0; q < 4; ++q) {
    int chunk = t + q * 256;
    srow[q] = chunk >> 3;
    scol[q] = (chunk & 7) * 4;
    bsrc[q] = Wb + (size_t)(n0 + srow[q]) * D_DIM + scol[q];
#if !USE_TDM
    a2src[q] = Ab + (size_t)srow[q] * D_DIM + scol[q];
#endif
  }

  v8f acc[4][2];
  {
    v8f z = {0.f,0.f,0.f,0.f,0.f,0.f,0.f,0.f};
    #pragma unroll
    for (int i = 0; i < 4; ++i)
      #pragma unroll
      for (int j = 0; j < 2; ++j) acc[i][j] = z;
  }

  v4f br[4];
#if !USE_TDM
  v4bf arr[4];
#endif

  // prologue: tile 0
#if USE_TDM
  if (wave == 0) tdm_load_tile_a(Ab, As[0]);
#else
  #pragma unroll
  for (int q = 0; q < 4; ++q) arr[q] = *(const v4bf*)(a2src[q]);
#endif
  #pragma unroll
  for (int q = 0; q < 4; ++q) br[q] = *(const v4f*)(bsrc[q]);
  #pragma unroll
  for (int q = 0; q < 4; ++q) cvt_store4(Bs[0] + srow[q] * 32 + scol[q], br[q]);
#if !USE_TDM
  #pragma unroll
  for (int q = 0; q < 4; ++q) *(v4bf*)(As[0] + srow[q] * 32 + scol[q]) = arr[q];
#endif
#if USE_TDM
  if (wave == 0) __builtin_amdgcn_s_wait_tensorcnt(0);
#endif

  const int KT = D_DIM / 32;
  int cur = 0;
  for (int kt = 0; kt < KT; ++kt) {
    __syncthreads();
    const bool more = (kt + 1 < KT);
    if (more) {
      int k0n = (kt + 1) * 32;
      #pragma unroll
      for (int q = 0; q < 4; ++q) br[q] = *(const v4f*)(bsrc[q] + k0n);
#if USE_TDM
      if (wave == 0) tdm_load_tile_a(Ab + k0n, As[cur ^ 1]);
#else
      #pragma unroll
      for (int q = 0; q < 4; ++q) arr[q] = *(const v4bf*)(a2src[q] + k0n);
#endif
    }
    mma_step(As[cur], Bs[cur], wm, wn, acc);
    if (more) {
      #pragma unroll
      for (int q = 0; q < 4; ++q) cvt_store4(Bs[cur ^ 1] + srow[q] * 32 + scol[q], br[q]);
#if !USE_TDM
      #pragma unroll
      for (int q = 0; q < 4; ++q) *(v4bf*)(As[cur ^ 1] + srow[q] * 32 + scol[q]) = arr[q];
#endif
#if USE_TDM
      if (wave == 0) __builtin_amdgcn_s_wait_tensorcnt(0);
#endif
    }
    cur ^= 1;
  }

  const int rhalf = (lane >> 4) * 8;
  #pragma unroll
  for (int i = 0; i < 4; ++i) {
    #pragma unroll
    for (int j = 0; j < 2; ++j) {
      int col = n0 + wn + j * 16 + (lane & 15);
      #pragma unroll
      for (int p = 0; p < 8; ++p) {
        int row = m0 + wm + i * 16 + rhalf + p;
        h_flat[((size_t)e * K_SEL + row) * C_DIM + col] = acc[i][j][p];
      }
    }
  }
}

// ---------------------------------------------------------------------------
// Launch: d_out = [h_flat E*K*C][local_indices E*K][weights_flat E*K][fanout N]
// d_ws   = [logits][weights][topk_idx][hidden bf16]  (~131 MB)
// ---------------------------------------------------------------------------
extern "C" void kernel_launch(void* const* d_in, const int* in_sizes, int n_in,
                              void* d_out, int out_size, void* d_ws, size_t ws_size,
                              hipStream_t stream) {
  (void)in_sizes; (void)n_in; (void)out_size; (void)ws_size;
  const float* x  = (const float*)d_in[0];
  const float* Wr = (const float*)d_in[1];
  const float* W1 = (const float*)d_in[2];
  const float* W2 = (const float*)d_in[3];

  float* out    = (float*)d_out;
  float* h_flat = out;
  float* out_li = h_flat + (size_t)E_EXP * K_SEL * C_DIM;
  float* out_w  = out_li + (size_t)E_EXP * K_SEL;
  float* fanout = out_w  + (size_t)E_EXP * K_SEL;

  char* ws = (char*)d_ws;
  float*  logits  = (float*)ws;  ws += (size_t)N_TOK * E_EXP * sizeof(float);
  float*  weights = (float*)ws;  ws += (size_t)N_TOK * E_EXP * sizeof(float);
  int*    topk    = (int*)ws;    ws += (size_t)E_EXP * K_SEL * sizeof(int);
  bf16_t* hidden  = (bf16_t*)ws;

  router_kernel<<<N_TOK, 256, 0, stream>>>(x, Wr, logits, weights, fanout);
  topk_kernel<<<E_EXP, 1024, 0, stream>>>(logits, weights, topk, out_li, out_w, fanout);
  expert_gemm1<<<dim3(D_DIM / 128, K_SEL / 128, E_EXP), 256, 0, stream>>>(x, W1, topk, hidden);
  expert_gemm2<<<dim3(C_DIM / 128, K_SEL / 128, E_EXP), 256, 0, stream>>>(hidden, W2, h_flat);
}
